// DotProductAttention_9543417332177
// MI455X (gfx1250) — matrix operand order repaired and checked by reading, not compiled, not run on hardware
//
#include <hip/hip_runtime.h>

typedef _Float16 f16;
typedef __attribute__((ext_vector_type(16))) _Float16 v16h;
typedef __attribute__((ext_vector_type(8)))  _Float16 v8h;
typedef __attribute__((ext_vector_type(4)))  _Float16 h4;
typedef __attribute__((ext_vector_type(2)))  _Float16 h2;
typedef __attribute__((ext_vector_type(8)))  float    v8f;
typedef __attribute__((ext_vector_type(4)))  float    f32x4;

typedef int v4i __attribute__((vector_size(16)));          // matches builtin param type
typedef __attribute__((address_space(1))) v4i gv4i;        // global b128
typedef __attribute__((address_space(3))) v4i lv4i;        // LDS b128
typedef __attribute__((address_space(3))) void lvoid;

#define B_   32
#define LQ_  2048
#define LK_  2048
#define D_   64
#define QTILE 128   // 8 waves * 16 q-rows
#define KC    64    // keys per chunk
#define NWAVE 8
#define SK 72       // halves per K-row in LDS  (64 + 8 pad; multiple of 8 keeps b128 16B-aligned)
#define SV 72       // halves per Vt-row in LDS
#define SP 72       // halves per P-row in LDS
#define QSCALE (0.125f * 1.44269504088896f)   // 1/sqrt(D) * log2(e): softmax in exp2 domain

#if defined(__has_builtin)
# if __has_builtin(__builtin_amdgcn_global_load_async_to_lds_b128)
#  define ASYNC_BUILTIN 1
# endif
# if __has_builtin(__builtin_amdgcn_s_wait_asynccnt)
#  define ASYNC_WAIT_BUILTIN 1
# endif
# if __has_builtin(__builtin_elementwise_maximumnum)
#  define H2MAX(a,b) __builtin_elementwise_maximumnum((a),(b))
# endif
#endif
#ifndef H2MAX
# define H2MAX(a,b) __builtin_elementwise_max((a),(b))
#endif

__device__ __forceinline__ void async_cp_b128(const f16* gsrc, f16* ldst) {
#if defined(ASYNC_BUILTIN)
  __builtin_amdgcn_global_load_async_to_lds_b128((gv4i*)(f16*)gsrc, (lv4i*)ldst, 0, 0);
#else
  unsigned lofs = (unsigned)(__SIZE_TYPE__)(lvoid*)ldst;          // 32-bit LDS offset
  unsigned long long ga = (unsigned long long)(__SIZE_TYPE__)gsrc;
  asm volatile("global_load_async_to_lds_b128 %0, %1, off"
               :: "v"(lofs), "v"(ga) : "memory");
#endif
}
__device__ __forceinline__ void wait_async0() {
#if defined(ASYNC_WAIT_BUILTIN)
  __builtin_amdgcn_s_wait_asynccnt(0);
#else
  asm volatile("s_wait_asynccnt 0x0" ::: "memory");
#endif
}

__device__ __forceinline__ v16h ld_frag(const f16* p0, const f16* p1) {
  v8h lo = *(const v8h*)p0;
  v8h hi = *(const v8h*)p1;
  return __builtin_shufflevector(lo, hi, 0,1,2,3,4,5,6,7,8,9,10,11,12,13,14,15);
}

// ---------------- pre-pass: K -> f16 row-major, V -> f16 transposed [b][d][key] ----------------
__global__ __launch_bounds__(256)
void preconvert_kernel(const float* __restrict__ K, const float* __restrict__ V,
                       f16* __restrict__ Kh, f16* __restrict__ Vth) {
  __shared__ __align__(16) f16 T[64 * 72];
  const int tid = threadIdx.x;
  const int bb  = blockIdx.y;
  const int k0  = blockIdx.x * 64;
  const f32x4* Kg = (const f32x4*)(K + (size_t)bb * LK_ * D_);
  const f32x4* Vg = (const f32x4*)(V + (size_t)bb * LK_ * D_);
  f16* KhB  = Kh  + (size_t)bb * LK_ * D_;
  f16* VthB = Vth + (size_t)bb * D_ * LK_;
#pragma unroll
  for (int i = 0; i < 4; ++i) {
    int f = tid + i * 256;               // 1024 float4s per 64x64 tile
    int key = f >> 4, d4 = (f & 15) << 2;
    f32x4 kx = Kg[(size_t)(k0 + key) * (D_ / 4) + (f & 15)];
    f32x4 vx = Vg[(size_t)(k0 + key) * (D_ / 4) + (f & 15)];
    h4 kh;
#pragma unroll
    for (int j = 0; j < 4; ++j) {
      kh[j] = (f16)kx[j];
      T[(d4 + j) * 72 + key] = (f16)vx[j];   // transpose via LDS
    }
    *(h4*)(KhB + (size_t)(k0 + key) * D_ + d4) = kh;
  }
  __syncthreads();
  const int d = tid >> 2, kk = (tid & 3) << 4;
#pragma unroll
  for (int c = 0; c < 2; ++c)
    *(v8h*)(VthB + (size_t)d * LK_ + k0 + kk + c * 8) =
        *(const v8h*)(T + d * 72 + kk + c * 8);
}

// ---------------- flash attention main kernel ----------------
template<bool PRE>
__global__ __launch_bounds__(256)
void DotProductAttention_9543417332177_kernel(const float* __restrict__ Q,
                                              const float* __restrict__ Kf,
                                              const float* __restrict__ Vf,
                                              float* __restrict__ O,
                                              const f16* __restrict__ Kh,
                                              const f16* __restrict__ Vth) {
  constexpr int NB = PRE ? 2 : 1;                      // double-buffer with async fills
  __shared__ __align__(16) f16 KshA[NB * KC * SK];     // K chunk(s), row-major [key][d]
  __shared__ __align__(16) f16 VtA [NB * D_ * SV];     // V chunk(s), transposed [d][key]
  __shared__ __align__(16) f16 Psh[NWAVE * 16 * SP];   // per-wave P scratch [row][key]

  const int tid  = threadIdx.x;
  const int wave = tid >> 5;
  const int lane = tid & 31;
  const int col  = lane & 15;
  const int hh   = lane >> 4;            // half-wave select (WMMA lane split)
  const int bb   = blockIdx.y;
  const int q0   = blockIdx.x * QTILE + wave * 16;

  // ---- Q into WMMA A-layout fragments; fold 1/sqrt(D)*log2(e) into Q ----
  const float* qptr = Q + ((size_t)bb * LQ_ + (q0 + col)) * D_;
  v16h qa[2];
#pragma unroll
  for (int f = 0; f < 2; ++f)
#pragma unroll
    for (int e = 0; e < 16; ++e) {
      int d = f * 32 + e + ((e >= 8) ? 8 : 0) + hh * 8;
      qa[f][e] = (f16)(qptr[d] * QSCALE);
    }

  v16h onesf;                    // row-sum of P via WMMA
#pragma unroll
  for (int e = 0; e < 16; ++e) onesf[e] = (f16)1.0f;

  v8f acc[4];
  v8f zero = {};
#pragma unroll
  for (int i = 0; i < 4; ++i) acc[i] = zero;
  float rmax[8], rsum[8];
#pragma unroll
  for (int r = 0; r < 8; ++r) { rmax[r] = -1e30f; rsum[r] = 0.f; }

  f16* psW = &Psh[wave * 16 * SP];
  const int fkey = tid >> 2;            // fill: K row / Vt row handled by this thread
  const int fd16 = (tid & 3) << 4;      // 16-half (32B) sub-block
  const f16*  KgH = Kh  + (size_t)bb * LK_ * D_;
  const f16*  VgH = Vth + (size_t)bb * D_ * LK_;
  const f32x4* Kg = (const f32x4*)(Kf + (size_t)bb * LK_ * D_);
  const f32x4* Vg = (const f32x4*)(Vf + (size_t)bb * LK_ * D_);

  // async fill of buffer pb with key chunk k0 (PRE path)
  auto async_fill = [&](int pb, int k0) {
    const f16* kp = KgH + (size_t)(k0 + fkey) * D_ + fd16;
    const f16* vp = VgH + (size_t)fkey * LK_ + k0 + fd16;
    f16* kd = KshA + pb * (KC * SK) + fkey * SK + fd16;
    f16* vd = VtA  + pb * (D_ * SV) + fkey * SV + fd16;
    async_cp_b128(kp,     kd);
    async_cp_b128(kp + 8, kd + 8);
    async_cp_b128(vp,     vd);
    async_cp_b128(vp + 8, vd + 8);
  };

  // fallback staging (f32 path)
  f32x4 kr32[4], vr32[4];
  auto preload32 = [&](int k0) {
#pragma unroll
    for (int i = 0; i < 4; ++i) {
      int f = tid + i * 256;
      kr32[i] = Kg[(size_t)(k0 + (f >> 4)) * (D_ / 4) + (f & 15)];
      vr32[i] = Vg[(size_t)(k0 + (f >> 4)) * (D_ / 4) + (f & 15)];
    }
  };
  auto fill32 = [&]() {
#pragma unroll
    for (int i = 0; i < 4; ++i) {
      int f = tid + i * 256;
      int key = f >> 4, d4 = (f & 15) << 2;
      f16* kp = &KshA[key * SK + d4];
#pragma unroll
      for (int j = 0; j < 4; ++j) {
        kp[j] = (f16)kr32[i][j];
        VtA[(d4 + j) * SV + key] = (f16)vr32[i][j];
      }
    }
  };

  // one flash step on the tile in (KshP, VtP)
  auto compute = [&](const f16* KshP, const f16* VtP) {
    // ---- S = Q K^T : four 16-key tiles, contraction over D=64 ----
    v8f s[4];
#pragma unroll
    for (int j = 0; j < 4; ++j) {
      v8f c = zero;
#pragma unroll
      for (int dc = 0; dc < 2; ++dc) {
        const f16* base = &KshP[(j * 16 + col) * SK + dc * 32 + hh * 8];
        v16h kb = ld_frag(base, base + 16);
        c = __builtin_amdgcn_wmma_f32_16x16x32_f16(false, qa[dc], false, kb,
                                                   (short)0, c, false, false);
      }
      s[j] = c;
    }

    // ---- packed half2 row-max reduction; hops batched across all 4 row-pairs ----
    h2 mx[4];
#pragma unroll
    for (int p = 0; p < 4; ++p) {
      int r0 = 2 * p, r1 = 2 * p + 1;
      mx[p][0] = (f16)fmaxf(fmaxf(s[0][r0], s[1][r0]), fmaxf(s[2][r0], s[3][r0]));
      mx[p][1] = (f16)fmaxf(fmaxf(s[0][r1], s[1][r1]), fmaxf(s[2][r1], s[3][r1]));
    }
#pragma unroll
    for (int msk = 1; msk <= 8; msk <<= 1) {
      int u[4];
#pragma unroll
      for (int p = 0; p < 4; ++p)
        u[p] = __shfl_xor(__builtin_bit_cast(int, mx[p]), msk, 32);
#pragma unroll
      for (int p = 0; p < 4; ++p)
        mx[p] = H2MAX(mx[p], __builtin_bit_cast(h2, u[p]));
    }
    float mn[8];
#pragma unroll
    for (int p = 0; p < 4; ++p) {
      mn[2 * p]     = fmaxf(rmax[2 * p],     (float)mx[p][0]);
      mn[2 * p + 1] = fmaxf(rmax[2 * p + 1], (float)mx[p][1]);
    }

    // ---- rescale + exp2 + stash P ----
#pragma unroll
    for (int r = 0; r < 8; ++r) {
      float corr = exp2f(rmax[r] - mn[r]);
      rmax[r] = mn[r];
      rsum[r] *= corr;
#pragma unroll
      for (int db = 0; db < 4; ++db) acc[db][r] *= corr;
      int row = r + (hh << 3);
#pragma unroll
      for (int j = 0; j < 4; ++j)
        psW[row * SP + j * 16 + col] = (f16)exp2f(s[j][r] - mn[r]);
    }

    asm volatile("" ::: "memory");   // per-wave LDS ops are in-order in HW

    // ---- P fragments, then row-sum (P x ones) and O += P V ----
    v16h pa[2];
#pragma unroll
    for (int c = 0; c < 2; ++c) {
      const f16* pbase = &psW[col * SP + c * 32 + hh * 8];   // A layout for P
      pa[c] = ld_frag(pbase, pbase + 16);
    }
    v8f csum = __builtin_amdgcn_wmma_f32_16x16x32_f16(false, pa[0], false, onesf,
                                                      (short)0, zero, false, false);
    csum = __builtin_amdgcn_wmma_f32_16x16x32_f16(false, pa[1], false, onesf,
                                                  (short)0, csum, false, false);
#pragma unroll
    for (int r = 0; r < 8; ++r) rsum[r] += csum[r];

#pragma unroll
    for (int db = 0; db < 4; ++db) {
#pragma unroll
      for (int c = 0; c < 2; ++c) {
        const f16* vbase = &VtP[(db * 16 + col) * SV + c * 32 + hh * 8];
        v16h vb = ld_frag(vbase, vbase + 16);
        acc[db] = __builtin_amdgcn_wmma_f32_16x16x32_f16(false, pa[c], false, vb,
                                                         (short)0, acc[db], false, false);
      }
    }
  };

  if constexpr (PRE) {
    // async double-buffered pipeline: one barrier per chunk
    async_fill(0, 0);
    for (int k0 = 0; k0 < LK_; k0 += KC) {
      int pb = (k0 >> 6) & 1;
      wait_async0();            // this wave's fill of pb complete
      __syncthreads();          // all waves' fills complete / prior reads done
      if (k0 + KC < LK_) async_fill(pb ^ 1, k0 + KC);   // overlaps compute
      compute(KshA + pb * (KC * SK), VtA + pb * (D_ * SV));
    }
  } else {
    preload32(0);
    for (int k0 = 0; k0 < LK_; k0 += KC) {
      fill32();
      __syncthreads();
      if (k0 + KC < LK_) preload32(k0 + KC);
      compute(KshA, VtA);
      __syncthreads();
    }
  }

  // ---- epilogue: normalize and store ----
#pragma unroll
  for (int r = 0; r < 8; ++r) {
    float inv = 1.f / rsum[r];
    int q = q0 + r + (hh << 3);
    float* op = O + ((size_t)bb * LQ_ + q) * D_ + col;
#pragma unroll
    for (int db = 0; db < 4; ++db)
      op[db * 16] = acc[db][r] * inv;
  }
}

extern "C" void kernel_launch(void* const* d_in, const int* in_sizes, int n_in,
                              void* d_out, int out_size, void* d_ws, size_t ws_size,
                              hipStream_t stream) {
  (void)in_sizes; (void)n_in; (void)out_size;
  const float* Q = (const float*)d_in[0];
  const float* K = (const float*)d_in[1];
  const float* V = (const float*)d_in[2];
  float* O = (float*)d_out;
  dim3 grid(LQ_ / QTILE, B_);
  dim3 block(256);

  const size_t elems = (size_t)B_ * LK_ * D_;
  const size_t need  = 2 * elems * sizeof(f16);   // Kh + Vth
  if (d_ws != nullptr && ws_size >= need) {
    f16* Kh  = (f16*)d_ws;
    f16* Vth = Kh + elems;
    preconvert_kernel<<<dim3(LK_ / 64, B_), block, 0, stream>>>(K, V, Kh, Vth);
    DotProductAttention_9543417332177_kernel<true>
        <<<grid, block, 0, stream>>>(Q, K, V, O, Kh, Vth);
  } else {
    DotProductAttention_9543417332177_kernel<false>
        <<<grid, block, 0, stream>>>(Q, K, V, O, nullptr, nullptr);
  }
}
